// CAModel_75307956568440
// MI455X (gfx1250) — matrix operand order repaired and checked by reading, not compiled, not run on hardware
//
#include <hip/hip_runtime.h>
#include <hip/hip_bf16.h>

typedef __attribute__((ext_vector_type(16))) _Float16 v16h;
typedef __attribute__((ext_vector_type(8)))  float    v8f;

#define HH 1024
#define WW 1024
#define HWSZ (HH * WW)

// Workspace layout (f16):
//   B1 fragments: [8][32 lanes][16]  -> 8192 B   (offset 0)
//   B2 fragments: [4][32 lanes][16]  -> 4096 B   (offset 8192)
#define WS_NEEDED 12288

// ---- gfx1250 async global->LDS path (guarded; falls back to sync staging) ----
#if defined(__has_builtin)
#  if __has_builtin(__builtin_amdgcn_global_load_async_to_lds_b32) && \
      __has_builtin(__builtin_amdgcn_s_wait_asynccnt)
#    define USE_ASYNC_LDS 1
#  endif
#endif
#ifndef USE_ASYNC_LDS
#  define USE_ASYNC_LDS 0
#endif

typedef __attribute__((address_space(1))) int g_int;
typedef __attribute__((address_space(3))) int l_int;

__device__ __forceinline__ void async_g2l_b32(const float* g, float* l) {
#if USE_ASYNC_LDS
    // Generic->AS1: global generic address == AS1 address.
    // Generic->AS3: LDS aperture address low 32 bits == LDS offset (ISA 10.2).
    __builtin_amdgcn_global_load_async_to_lds_b32(
        (g_int*)(uintptr_t)g,
        (l_int*)(uint32_t)(uintptr_t)l,
        0, 0);
#else
    (void)g; (void)l;
#endif
}

// ---------------------------------------------------------------------------
// Prep kernel: fold w1 -> weff (128x32), convert to f16, and store B-matrix
// fragments in the exact per-lane WMMA register layout:
//   B1 (32x16 f16): lanes 0-15 hold K=0..7,16..23 (col N=lane),
//                         lanes 16-31 hold K=8..15,24..31 (col N=lane-16); B2: half h holds K=16h..16h+15.
// ---------------------------------------------------------------------------
__global__ __launch_bounds__(256) void ca_prep_weights(
    const float* __restrict__ w1, const float* __restrict__ w2,
    _Float16* __restrict__ wsf)
{
    const int t    = threadIdx.x;
    const int lane = t & 31;
    const int nl   = lane & 15;
    const bool hi  = lane >= 16;
    const int frag = t >> 5; // 0..7

    // GEMM1 fragment: weff[j][k<16] = w1[j][k] + w1[j][k+16]; weff[j][16+c] = w1[j][32+c]
    {
        const int j  = 16 * frag + nl;
        const float* wr = w1 + j * 48;
        const int kb = hi ? 8 : 0;
        _Float16* dst = wsf + (frag * 32 + lane) * 16;
#pragma unroll
        for (int e = 0; e < 16; ++e) {
            const int k = (e < 8) ? (kb + e) : (8 + kb + e);
            float v = (k < 16) ? (wr[k] + wr[k + 16]) : wr[k + 16];
            dst[e] = (_Float16)v;
        }
    }
    // GEMM2 fragments: B2[t] column c = lane%16, K = 32t + (hi?16:0) + e
    if (frag < 4) {
        const float* wr = w2 + nl * 128 + 32 * frag + (hi ? 16 : 0);
        _Float16* dst = wsf + (256 + frag * 32 + lane) * 16;
#pragma unroll
        for (int e = 0; e < 16; ++e)
            dst[e] = (_Float16)wr[e];
    }
}

// ---------------------------------------------------------------------------
// Main fused kernel. PACKED=true loads weight fragments from workspace with
// coalesced 32B vector loads; PACKED=false builds them from w1/w2 (fallback).
// ---------------------------------------------------------------------------
template <bool PACKED>
__global__ __launch_bounds__(256) void ca_nca_wmma_kernel(
    const float* __restrict__ in,    // (16, H, W)
    const float* __restrict__ stoch, // (H, W)
    const _Float16* __restrict__ wpack,
    const float* __restrict__ w1,    // (128, 48)
    const float* __restrict__ b1,    // (128,)
    const float* __restrict__ w2,    // (16, 128)
    const float* __restrict__ b2,    // (16,)
    float* __restrict__ out)         // (16, H, W)
{
    __shared__ __align__(16) float t_in[16][18][20];        // raw input tile + halo
    __shared__ __align__(16) _Float16 h_lds[8][16][136];    // per-wave hidden acts
    __shared__ float am[8][16];                             // alive per pixel
    __shared__ float mk[8][16];                             // stochastic mask

    const int tid  = threadIdx.x;
    const int wave = tid >> 5;
    const int lane = tid & 31;
    const int nl   = lane & 15;
    const bool hi  = lane >= 16;
    const int tx   = blockIdx.x * 16;
    const int ty   = blockIdx.y * 16;

    // ---- cooperative halo tile staging: 324 (r,col) positions x 16 channels ----
#pragma unroll
    for (int pp = 0; pp < 2; ++pp) {
        const int p = tid + pp * 256;
        if (p < 18 * 18) {
            const int r   = p / 18;        // constant division (mul/shift)
            const int col = p - r * 18;
            const int gy  = ty + r - 1;
            const int gx  = tx + col - 1;
            float* lp = &t_in[0][r][col];  // channel stride = 360 floats
            if ((unsigned)gy < (unsigned)HH && (unsigned)gx < (unsigned)WW) {
                const float* gp = in + gy * WW + gx;
#if USE_ASYNC_LDS
#pragma unroll
                for (int c = 0; c < 16; ++c)
                    async_g2l_b32(gp + c * HWSZ, lp + c * (18 * 20));
#else
#pragma unroll
                for (int c = 0; c < 16; ++c)
                    lp[c * (18 * 20)] = gp[c * HWSZ];
#endif
            } else {
#pragma unroll
                for (int c = 0; c < 16; ++c)
                    lp[c * (18 * 20)] = 0.0f;
            }
        }
    }

    // ---- weight fragments into registers (overlaps with async staging) ----
    v16h B1f[8];
    v16h B2f[4];
    float bias1[8];
    if (PACKED) {
        const v16h* wf = (const v16h*)wpack;
#pragma unroll
        for (int n = 0; n < 8; ++n) B1f[n] = wf[n * 32 + lane];
#pragma unroll
        for (int t = 0; t < 4; ++t) B2f[t] = wf[256 + t * 32 + lane];
    } else {
        const int kb = hi ? 8 : 0;
#pragma unroll
        for (int n = 0; n < 8; ++n) {
            const float* wr = w1 + (16 * n + nl) * 48;
#pragma unroll
            for (int e = 0; e < 16; ++e) {
                const int k = (e < 8) ? (kb + e) : (8 + kb + e);
                float v = (k < 16) ? (wr[k] + wr[k + 16]) : wr[k + 16];
                B1f[n][e] = (_Float16)v;
            }
        }
#pragma unroll
        for (int t = 0; t < 4; ++t) {
            const float* wr = w2 + nl * 128 + 32 * t + (hi ? 16 : 0);
#pragma unroll
            for (int e = 0; e < 16; ++e)
                B2f[t][e] = (_Float16)wr[e];
        }
    }
#pragma unroll
    for (int n = 0; n < 8; ++n) bias1[n] = b1[16 * n + nl];
    const float bias2 = b2[nl];

#if USE_ASYNC_LDS
    __builtin_amdgcn_s_wait_asynccnt(0); // staging landed in LDS
#endif
    __syncthreads();

    const int chb = hi ? 8 : 0; // per-lane-half K base (A layout) = channel base

    for (int s = 0; s < 2; ++s) {
        const int sr = 2 * wave + s; // strip row within tile
        const int M  = nl;           // this lane's A-row = pixel column

        // A1 (16-bit A layout): e0..7 -> K=chb+e (sx), e8..15 -> K=chb+8+e (idc)
        v16h A1;
#pragma unroll
        for (int q = 0; q < 8; ++q) {
            const int c = chb + q;
            const float(*tc)[20] = t_in[c];
            float d0 = tc[sr][M]     - tc[sr][M + 2];
            float d1 = tc[sr + 1][M] - tc[sr + 1][M + 2];
            float d2 = tc[sr + 2][M] - tc[sr + 2][M + 2];
            float sx = d0 + 2.0f * d1 + d2;
            sx = fminf(fmaxf(sx, 0.0f), 1.0f);
            float id = tc[sr + 1][M + 1];
            id = fminf(fmaxf(id, 0.0f), 1.0f);
            A1[q]     = (_Float16)sx;
            A1[q + 8] = (_Float16)id;
        }

        // alive (lanes 0-15) / stochastic mask (lanes 16-31)
        if (!hi) {
            const float(*t3)[20] = t_in[3];
            float mx = t3[sr][M];
#pragma unroll
            for (int dy = 0; dy < 3; ++dy)
#pragma unroll
                for (int dx = 0; dx < 3; ++dx)
                    mx = fmaxf(mx, t3[sr + dy][M + dx]);
            am[wave][M] = (mx > 0.1f) ? 1.0f : 0.0f;
        } else {
            float sv = stoch[(ty + sr) * WW + tx + M];
            mk[wave][M] = (sv < 0.5f) ? 1.0f : 0.0f;
        }

        // GEMM1: h = relu(A1 @ weffT + b1), 8 N-tiles
#pragma unroll
        for (int n = 0; n < 8; ++n) {
            v8f c = {};
            c = __builtin_amdgcn_wmma_f32_16x16x32_f16(
                    false, A1, false, B1f[n], (short)0, c, false, false);
            const int j = 16 * n + nl;
#pragma unroll
            for (int r = 0; r < 8; ++r) {
                float hv = fmaxf(c[r] + bias1[n], 0.0f);
                const int Mr = r + (hi ? 8 : 0); // C/D layout: M = r + 8*(lane/16)
                h_lds[wave][Mr][j] = (_Float16)hv;
            }
        }

        __syncthreads(); // order cross-lane LDS stores vs reads

        // GEMM2: x = h @ w2T + b2, K=128 as 4 accumulating WMMAs
        v8f acc = {};
#pragma unroll
        for (int t = 0; t < 4; ++t) {
            const int kb = 32 * t + 2 * chb;
            v16h A2;
#pragma unroll
            for (int e = 0; e < 8; ++e) A2[e]     = h_lds[wave][M][kb + e];
#pragma unroll
            for (int e = 0; e < 8; ++e) A2[e + 8] = h_lds[wave][M][kb + 8 + e];
            acc = __builtin_amdgcn_wmma_f32_16x16x32_f16(
                      false, A2, false, B2f[t], (short)0, acc, false, false);
        }

        // finalize: out = clip((x*mask + idc)*alive, 0, 1)
        const int ch = nl;
        const int gy = ty + sr;
#pragma unroll
        for (int r = 0; r < 8; ++r) {
            const int Mr = r + (hi ? 8 : 0);
            float x = acc[r] + bias2;
            float idv = t_in[ch][sr + 1][Mr + 1];
            idv = fminf(fmaxf(idv, 0.0f), 1.0f);
            float v = (x * mk[wave][Mr] + idv) * am[wave][Mr];
            v = fminf(fmaxf(v, 0.0f), 1.0f);
            out[ch * HWSZ + gy * WW + (tx + Mr)] = v;
        }

        __syncthreads(); // protect per-wave LDS buffers before next strip
    }
}

extern "C" void kernel_launch(void* const* d_in, const int* in_sizes, int n_in,
                              void* d_out, int out_size, void* d_ws, size_t ws_size,
                              hipStream_t stream) {
    (void)in_sizes; (void)n_in; (void)out_size;
    const float* in    = (const float*)d_in[0];
    const float* stoch = (const float*)d_in[1];
    const float* w1    = (const float*)d_in[2];
    const float* b1    = (const float*)d_in[3];
    const float* w2    = (const float*)d_in[4];
    const float* b2    = (const float*)d_in[5];
    float* out = (float*)d_out;

    dim3 grid(WW / 16, HH / 16); // 64 x 64 tiles of 16x16 pixels

    if (ws_size >= (size_t)WS_NEEDED && d_ws != nullptr) {
        _Float16* wsf = (_Float16*)d_ws;
        ca_prep_weights<<<1, 256, 0, stream>>>(w1, w2, wsf);
        ca_nca_wmma_kernel<true><<<grid, 256, 0, stream>>>(
            in, stoch, (const _Float16*)wsf, w1, b1, w2, b2, out);
    } else {
        ca_nca_wmma_kernel<false><<<grid, 256, 0, stream>>>(
            in, stoch, nullptr, w1, b1, w2, b2, out);
    }
}
